// LSTMModel_35373350650291
// MI455X (gfx1250) — matrix operand order repaired and checked by reading, not compiled, not run on hardware
//
#include <hip/hip_runtime.h>
#include <math.h>

#define B_   1024
#define T_   256
#define D_   128
#define H_   64
#define NKC  6    // K chunks of 32 over K = D+H = 192
#define NNT  16   // N tiles of 16 over 4H = 256

typedef __attribute__((ext_vector_type(16))) __bf16 v16bf;
typedef __attribute__((ext_vector_type(8)))  float  v8f;

union AFrag {
  v16bf v;
  __bf16 e[16];
  unsigned u[8];
};

__device__ __forceinline__ float fast_rcp(float x) { return __builtin_amdgcn_rcpf(x); }
__device__ __forceinline__ float sigf(float x) {
  return fast_rcp(1.0f + __expf(-x));                    // v_exp + v_add + v_rcp
}
__device__ __forceinline__ float tanh_f(float x) {
  return __fmaf_rn(2.0f, fast_rcp(1.0f + __expf(-2.0f * x)), -1.0f);
}

// Dynamic LDS layout (bytes):
//   [0, 98304)          : weight B-fragments, bf16, 96 frags * 32 lanes * 8 dwords
//   [98304, 100352)     : h_buf  16x64 bf16 (row-major, rows = batch-in-tile)
//   [100352, 104448)    : h_f32  16x64 f32
#define SMEM_BYTES 104448

__global__ __launch_bounds__(128)
void lstm_fused_kernel(const float* __restrict__ x,
                       const float* __restrict__ W_ih,
                       const float* __restrict__ W_hh,
                       const float* __restrict__ b_ih,
                       const float* __restrict__ b_hh,
                       const float* __restrict__ W1,
                       const float* __restrict__ b1,
                       const float* __restrict__ W2,
                       const float* __restrict__ b2,
                       float* __restrict__ out)
{
  extern __shared__ __align__(16) unsigned char smem_raw[];
  unsigned* lds_w = (unsigned*)smem_raw;                   // 96 KB
  __bf16*   h_buf = (__bf16*)(smem_raw + 98304);           // 2 KB
  float*    h_f32 = (float*)(smem_raw + 100352);           // 4 KB

  const int tid  = threadIdx.x;
  const int lane = tid & 31;
  const int cg   = tid >> 5;        // wave id = hidden column group 0..3
  const int ln   = lane & 15;
  const int half = lane >> 4;
  const int row_base = blockIdx.x * 16;

  // ---- one-time setup: weights -> LDS as bf16 B-fragments ----
  // B-matrix (32x16, KxN) lane layout: n = lane%16 ; k = 16*(e>=8) + 8*half + e%8
  for (int frag = cg; frag < NNT * NKC; frag += 4) {
    const int nt = frag / NKC;
    const int kc = frag % NKC;
    const int n  = nt * 16 + ln;
    AFrag w;
#pragma unroll
    for (int j = 0; j < 8; ++j) {
      const int k = kc * 32 + half * 8 + 2 * j + (j >= 4 ? 8 : 0);   // contiguous pair (k, k+1)
      float a, b;
      if (k < D_) { a = W_ih[n * D_ + k];        b = W_ih[n * D_ + k + 1]; }
      else        { a = W_hh[n * H_ + (k - D_)]; b = W_hh[n * H_ + (k - D_) + 1]; }
      w.e[2 * j]     = (__bf16)a;
      w.e[2 * j + 1] = (__bf16)b;
    }
    unsigned* dst = &lds_w[((unsigned)frag * 32u + (unsigned)lane) * 8u];
    *(uint4*)(dst)     = make_uint4(w.u[0], w.u[1], w.u[2], w.u[3]);
    *(uint4*)(dst + 4) = make_uint4(w.u[4], w.u[5], w.u[6], w.u[7]);
  }
  // zero h_buf (h0 = 0)
  for (int i = tid; i < 16 * 64; i += 128) h_buf[i] = (__bf16)0.0f;

  // per-lane bias for the 4 gate tiles this wave owns: columns (g*4+cg)*16 + ln
  float bias_r[4];
#pragma unroll
  for (int g = 0; g < 4; ++g) {
    const int col = (g * 4 + cg) * 16 + ln;
    bias_r[g] = b_ih[col] + b_hh[col];
  }
  __syncthreads();

  // cell / hidden state for this wave's 16x16 tile (C-layout: lane->n, vgpr r->m)
  float c_st[8], h_st[8];
#pragma unroll
  for (int r = 0; r < 8; ++r) { c_st[r] = 0.f; h_st[r] = 0.f; }

  const float* xrow = x + (size_t)(row_base + ln) * T_ * D_;

  for (int t = 0; t < T_; ++t) {
    const float* xp = xrow + (size_t)t * D_;

    // ---- A fragments from x_t (global f32 -> bf16 via hw cvt), A-layout:
    //      m = lane%16, element e -> k = 32*kc + 16*(e>=8) + 8*half + e%8
    AFrag ax[4];
#pragma unroll
    for (int kc = 0; kc < 4; ++kc) {
      const int k0 = kc * 32 + half * 8;
      float4 p0 = *(const float4*)(xp + k0);
      float4 p1 = *(const float4*)(xp + k0 + 4);
      float4 p2 = *(const float4*)(xp + k0 + 16);
      float4 p3 = *(const float4*)(xp + k0 + 20);
      ax[kc].e[0]  = (__bf16)p0.x;  ax[kc].e[1]  = (__bf16)p0.y;
      ax[kc].e[2]  = (__bf16)p0.z;  ax[kc].e[3]  = (__bf16)p0.w;
      ax[kc].e[4]  = (__bf16)p1.x;  ax[kc].e[5]  = (__bf16)p1.y;
      ax[kc].e[6]  = (__bf16)p1.z;  ax[kc].e[7]  = (__bf16)p1.w;
      ax[kc].e[8]  = (__bf16)p2.x;  ax[kc].e[9]  = (__bf16)p2.y;
      ax[kc].e[10] = (__bf16)p2.z;  ax[kc].e[11] = (__bf16)p2.w;
      ax[kc].e[12] = (__bf16)p3.x;  ax[kc].e[13] = (__bf16)p3.y;
      ax[kc].e[14] = (__bf16)p3.z;  ax[kc].e[15] = (__bf16)p3.w;
      // prefetch next timestep's x (lowers to global_prefetch_b8)
      if (t + 1 < T_) __builtin_prefetch(xp + D_ + k0, 0, 0);
    }

    // ---- A fragments from h_{t-1} (LDS bf16) ----
    AFrag ah[2];
#pragma unroll
    for (int c = 0; c < 2; ++c) {
      const __bf16* hp = &h_buf[ln * 64 + c * 32 + half * 8];
      uint4 q0 = *(const uint4*)hp;          // elements e = 0..7
      uint4 q1 = *(const uint4*)(hp + 16);   // elements e = 8..15
      ah[c].u[0] = q0.x; ah[c].u[1] = q0.y; ah[c].u[2] = q0.z; ah[c].u[3] = q0.w;
      ah[c].u[4] = q1.x; ah[c].u[5] = q1.y; ah[c].u[6] = q1.z; ah[c].u[7] = q1.w;
    }

    // ---- gate GEMMs: 4 tiles x 6 WMMAs (K = 192), bias pre-loaded in C ----
    v8f acc[4];
#pragma unroll
    for (int g = 0; g < 4; ++g) {
      const float bb = bias_r[g];
      v8f a = {bb, bb, bb, bb, bb, bb, bb, bb};
      const int nt = g * 4 + cg;
#pragma unroll
      for (int kc = 0; kc < NKC; ++kc) {
        const unsigned* wp = &lds_w[((unsigned)(nt * NKC + kc) * 32u + (unsigned)lane) * 8u];
        AFrag wf;
        *(uint4*)&wf.u[0] = *(const uint4*)wp;
        *(uint4*)&wf.u[4] = *(const uint4*)(wp + 4);
        a = __builtin_amdgcn_wmma_f32_16x16x32_bf16(false,
                (kc < 4) ? ax[kc].v : ah[kc - 4].v, false, wf.v,
                (short)0, a, false, false);
      }
      acc[g] = a;
    }

    // ---- elementwise LSTM cell update (all four gates share (m,n) layout) ----
#pragma unroll
    for (int r = 0; r < 8; ++r) {
      float iv = sigf  (acc[0][r]);
      float fv = sigf  (acc[1][r]);
      float gv = tanh_f(acc[2][r]);
      float ov = sigf  (acc[3][r]);
      float cv = __fmaf_rn(fv, c_st[r], iv * gv);
      c_st[r] = cv;
      h_st[r] = ov * tanh_f(cv);
    }

    __syncthreads();   // previous h_buf reads complete
#pragma unroll
    for (int r = 0; r < 8; ++r) {
      const int m = r + 8 * half;
      h_buf[m * 64 + cg * 16 + ln] = (__bf16)h_st[r];
    }
    __syncthreads();   // h_buf ready for next step
  }

  // ---- final h (f32) -> LDS, then tiny MLP head ----
#pragma unroll
  for (int r = 0; r < 8; ++r) {
    const int m = r + 8 * half;
    h_f32[m * 64 + cg * 16 + ln] = h_st[r];
  }
  __syncthreads();

  if (tid < 16) {
    const int m = tid;
    float acc2 = b2[0];
    for (int u = 0; u < 32; ++u) {
      float a = b1[u];
      const float* w1r = W1 + u * 64;
#pragma unroll 8
      for (int k = 0; k < 64; ++k) a = __fmaf_rn(h_f32[m * 64 + k], w1r[k], a);
      a = fmaxf(a, 0.f);
      acc2 = __fmaf_rn(a, W2[u], acc2);
    }
    out[row_base + m] = fast_rcp(1.0f + __expf(-acc2));
  }
}

extern "C" void kernel_launch(void* const* d_in, const int* in_sizes, int n_in,
                              void* d_out, int out_size, void* d_ws, size_t ws_size,
                              hipStream_t stream) {
  (void)in_sizes; (void)n_in; (void)d_ws; (void)ws_size; (void)out_size;
  const float* x    = (const float*)d_in[0];
  const float* W_ih = (const float*)d_in[1];
  const float* W_hh = (const float*)d_in[2];
  const float* b_ih = (const float*)d_in[3];
  const float* b_hh = (const float*)d_in[4];
  const float* W1   = (const float*)d_in[5];
  const float* b1   = (const float*)d_in[6];
  const float* W2   = (const float*)d_in[7];
  const float* b2   = (const float*)d_in[8];
  float* out = (float*)d_out;

  hipFuncSetAttribute((const void*)lstm_fused_kernel,
                      hipFuncAttributeMaxDynamicSharedMemorySize, SMEM_BYTES);

  lstm_fused_kernel<<<B_ / 16, 128, SMEM_BYTES, stream>>>(
      x, W_ih, W_hh, b_ih, b_hh, W1, b1, W2, b2, out);
}
